// MultiHeadedAttention_54597624266793
// MI455X (gfx1250) — matrix operand order repaired and checked by reading, not compiled, not run on hardware
//
#include <hip/hip_runtime.h>

typedef __attribute__((ext_vector_type(16))) _Float16 v16h;
typedef __attribute__((ext_vector_type(8)))  float    v8f;

#define M_TOTAL 16384   // bt*n = 16*1024 rows
#define C_DIM   512     // model dim
#define S_LEN   2048    // tokens per attention group (t*h2*w2)
#define CH      64      // head dim
#define NGROUP  64      // b(2) * windows(4) * heads(8)

// ---- gfx1250 async global->LDS (builtin confirmed present by probe) ----
#if defined(__has_builtin)
#if __has_builtin(__builtin_amdgcn_global_load_async_to_lds_b128) && \
    __has_builtin(__builtin_amdgcn_s_wait_asynccnt)
#define USE_ASYNC_LDS 1
// exact parameter type per hipcc diagnostic: int __vector_size__(16) __device__*
typedef int vsi4 __attribute__((vector_size(16)));
typedef __attribute__((address_space(1))) vsi4* gptr_b128;
typedef __attribute__((address_space(3))) vsi4* lptr_b128;
#endif
#endif

static __device__ __forceinline__ v8f wmma_f16(v16h a, v16h b, v8f c) {
    // D = A(16x32 f16) * B(32x16 f16) + C(16x16 f32)
    return __builtin_amdgcn_wmma_f32_16x16x32_f16(
        /*neg_a=*/false, a, /*neg_b=*/false, b,
        /*c_mod=*/(short)0, c, /*reuse_a=*/false, /*reuse_b=*/false);
}

// Map flat row m in [0, 16384) onto (window-group bw in [0,8), seq pos s in [0,2048))
static __device__ __forceinline__ void row_to_window(int m, int& bw, int& s) {
    int bt = m >> 10;
    int n  = m & 1023;
    int b  = bt >> 3;
    int tt = bt & 7;
    int hh = n >> 5;
    int ww = n & 31;
    int win = ((hh >> 4) << 1) | (ww >> 4);
    bw = (b << 2) | win;
    s  = (tt << 8) | ((hh & 15) << 4) | (ww & 15);
}

// ---------------------------------------------------------------------------
// Kernel 0: one-shot f32 -> f16 conversion (x and weights), 4 elems/thread
// ---------------------------------------------------------------------------
__global__ __launch_bounds__(256)
void cvt_f32_to_f16_kernel(const float* __restrict__ src,
                           _Float16* __restrict__ dst, int n) {
    int i = (blockIdx.x * 256 + threadIdx.x) * 4;
    if (i < n) {
        float4 v = *(const float4*)(src + i);
        dst[i + 0] = (_Float16)v.x;
        dst[i + 1] = (_Float16)v.y;
        dst[i + 2] = (_Float16)v.z;
        dst[i + 3] = (_Float16)v.w;
    }
}

// ---------------------------------------------------------------------------
// Kernel 1: fused QKV projection from pre-converted f16 operands.
// One wave computes a 32(M) x 32(N) tile of q,k,v (12 WMMAs per K-step).
// Q,K stored row-major [g][S][CH]; V stored transposed [g][CH][S].
// ---------------------------------------------------------------------------
__global__ __launch_bounds__(32)
void qkv_proj_kernel(const _Float16* __restrict__ Xh,
                     const _Float16* __restrict__ Whq, const float* __restrict__ bq,
                     const _Float16* __restrict__ Whk, const float* __restrict__ bk,
                     const _Float16* __restrict__ Whv, const float* __restrict__ bv,
                     _Float16* __restrict__ Qw, _Float16* __restrict__ Kw,
                     _Float16* __restrict__ Vt) {
    const int lane = threadIdx.x;
    const int h    = lane >> 4;
    const int l15  = lane & 15;
    const int m0   = blockIdx.x << 5;   // 32 rows
    const int n0   = blockIdx.y << 5;   // 32 cols

    v8f cq[2][2], ck[2][2], cv[2][2];   // [mt][nt]
    #pragma unroll
    for (int nt = 0; nt < 2; ++nt) {
        float vq = bq[n0 + (nt << 4) + l15];
        float vk = bk[n0 + (nt << 4) + l15];
        float vv = bv[n0 + (nt << 4) + l15];
        #pragma unroll
        for (int mt = 0; mt < 2; ++mt)
            #pragma unroll
            for (int r = 0; r < 8; ++r) {
                cq[mt][nt][r] = vq; ck[mt][nt][r] = vk; cv[mt][nt][r] = vv;
            }
    }

    for (int k0 = 0; k0 < C_DIM; k0 += 32) {
        v16h a[2];
        #pragma unroll
        for (int mt = 0; mt < 2; ++mt) {
            const _Float16* xr = Xh + (size_t)(m0 + (mt << 4) + l15) * C_DIM + k0;
            #pragma unroll
            for (int j = 0; j < 8; ++j) {
                a[mt][j]     = xr[(h << 4) + j];
                a[mt][8 + j] = xr[8 + (h << 4) + j];
            }
        }
        #pragma unroll
        for (int nt = 0; nt < 2; ++nt) {
            const size_t wo = (size_t)(n0 + (nt << 4) + l15) * C_DIM + k0 + (h << 4);
            v16h bqm, bkm, bvm;
            #pragma unroll
            for (int j = 0; j < 16; ++j) {
                bqm[j] = Whq[wo + j];
                bkm[j] = Whk[wo + j];
                bvm[j] = Whv[wo + j];
            }
            #pragma unroll
            for (int mt = 0; mt < 2; ++mt) {
                cq[mt][nt] = wmma_f16(a[mt], bqm, cq[mt][nt]);
                ck[mt][nt] = wmma_f16(a[mt], bkm, ck[mt][nt]);
                cv[mt][nt] = wmma_f16(a[mt], bvm, cv[mt][nt]);
            }
        }
    }

    // Scatter into windowed layouts (D layout: M = r + 8h, N = l15)
    #pragma unroll
    for (int nt = 0; nt < 2; ++nt) {
        const int col  = n0 + (nt << 4) + l15;
        const int head = col >> 6;
        const int cc   = col & 63;
        #pragma unroll
        for (int mt = 0; mt < 2; ++mt)
            #pragma unroll
            for (int r = 0; r < 8; ++r) {
                int m = m0 + (mt << 4) + r + (h << 3);
                int bw, s;
                row_to_window(m, bw, s);
                int gh = (bw << 3) | head;
                Qw[((size_t)gh * S_LEN + s) * CH + cc] = (_Float16)cq[mt][nt][r];
                Kw[((size_t)gh * S_LEN + s) * CH + cc] = (_Float16)ck[mt][nt][r];
                Vt[(size_t)gh * (CH * S_LEN) + (size_t)cc * S_LEN + s] =
                    (_Float16)cv[mt][nt][r];
            }
    }
}

// ---------------------------------------------------------------------------
// K/V tile staging: 8 KB per tile pair; threads 0-127 copy K, 128-255 copy V.
// Async path issues 2 x b128 per thread into LDS tracked by ASYNCcnt.
// ---------------------------------------------------------------------------
static __device__ __forceinline__ void stage_tiles(
    int tid, const _Float16* __restrict__ Kg, const _Float16* __restrict__ Vg,
    int jb, _Float16* dK, _Float16* dV) {
#ifdef USE_ASYNC_LDS
    if (tid < 128) {
        const _Float16* s = Kg + (size_t)jb * CH + tid * 16;
        _Float16* d = dK + tid * 16;
        __builtin_amdgcn_global_load_async_to_lds_b128(
            (gptr_b128)s, (lptr_b128)d, 0, 0);
        __builtin_amdgcn_global_load_async_to_lds_b128(
            (gptr_b128)s, (lptr_b128)d, 16, 0);
    } else {
        int idx = tid - 128;            // 0..127
        int ch  = idx >> 1;
        int seg = idx & 1;
        const _Float16* s = Vg + (size_t)ch * S_LEN + jb + seg * 16;
        _Float16* d = dV + ch * 32 + seg * 16;
        __builtin_amdgcn_global_load_async_to_lds_b128(
            (gptr_b128)s, (lptr_b128)d, 0, 0);
        __builtin_amdgcn_global_load_async_to_lds_b128(
            (gptr_b128)s, (lptr_b128)d, 16, 0);
    }
#else
    if (tid < 128) {
        const uint4* gk = (const uint4*)(Kg + (size_t)jb * CH);
        uint4* lk = (uint4*)dK;
        lk[tid * 2]     = gk[tid * 2];
        lk[tid * 2 + 1] = gk[tid * 2 + 1];
    } else {
        int idx = tid - 128;
        int ch  = idx >> 1;
        int seg = idx & 1;
        const uint4* gv = (const uint4*)(Vg + (size_t)ch * S_LEN + jb + seg * 16);
        uint4* lv = (uint4*)(dV + ch * 32 + seg * 16);
        lv[0] = gv[0];
        lv[1] = gv[1];
    }
#endif
}

// ---------------------------------------------------------------------------
// Kernel 2: flash attention. Block = 8 waves; each wave owns 32 query rows.
// Double-buffered LDS K/V tiles (async copy overlapped with compute);
// softmax statistics kept in registers (row m = r+8h is owned by one
// 16-lane half, so row reductions are 4 shfl_xor ops; alpha/l line up
// with the accumulator's r index by construction).
// ---------------------------------------------------------------------------
__global__ __launch_bounds__(256)
void attn_kernel(const _Float16* __restrict__ Qw, const _Float16* __restrict__ Kw,
                 const _Float16* __restrict__ Vt, _Float16* __restrict__ Att) {
    __shared__ __align__(16) _Float16 sK[2][32 * 64];    // keys tile, row-major
    __shared__ __align__(16) _Float16 sV[2][64 * 32];    // V^T tile: [ch][key]
    __shared__ __align__(16) _Float16 sP[8][32 * 32];    // per-wave probs

    const int tid  = threadIdx.x;
    const int wv   = tid >> 5;
    const int lane = tid & 31;
    const int h    = lane >> 4;
    const int l15  = lane & 15;
    const int g    = blockIdx.y;                 // group 0..63
    const int q0   = (blockIdx.x << 8) + (wv << 5);   // this wave's 32 q rows

    const _Float16* Qg = Qw + (size_t)g * (S_LEN * CH);
    const _Float16* Kg = Kw + (size_t)g * (S_LEN * CH);
    const _Float16* Vg = Vt + (size_t)g * (CH * S_LEN);

    // Q block (32 x 64) as 2x2 A tiles, kept in registers.
    v16h aq[2][2];   // [mt][kc]
    #pragma unroll
    for (int mt = 0; mt < 2; ++mt)
        #pragma unroll
        for (int kc = 0; kc < 2; ++kc) {
            const _Float16* qr = Qg + (size_t)(q0 + (mt << 4) + l15) * CH + (kc << 5);
            #pragma unroll
            for (int j = 0; j < 8; ++j) {
                aq[mt][kc][j]     = qr[(h << 4) + j];
                aq[mt][kc][8 + j] = qr[8 + (h << 4) + j];
            }
        }

    v8f acc[2][4];                                // [mt][channel tile]
    float m_run[2][8], l_run[2][8];               // per-row stats (row r+8h)
    #pragma unroll
    for (int mt = 0; mt < 2; ++mt) {
        #pragma unroll
        for (int t = 0; t < 4; ++t)
            #pragma unroll
            for (int r = 0; r < 8; ++r) acc[mt][t][r] = 0.0f;
        #pragma unroll
        for (int r = 0; r < 8; ++r) { m_run[mt][r] = -1e30f; l_run[mt][r] = 0.0f; }
    }

    const int NSTEP = S_LEN / 32;   // 64
    stage_tiles(tid, Kg, Vg, 0, sK[0], sV[0]);

    for (int it = 0; it < NSTEP; ++it) {
        const int cur = it & 1;
        const int jb  = it << 5;
        __syncthreads();   // all waves done reading buffer cur^1
        if (it + 1 < NSTEP)
            stage_tiles(tid, Kg, Vg, jb + 32, sK[cur ^ 1], sV[cur ^ 1]);
        // L2 prefetch two tiles ahead
        if (jb + 64 < S_LEN) {
            if (tid < 64)
                __builtin_prefetch(Kg + (size_t)(jb + 64) * CH + tid * 64, 0, 1);
            else if (tid < 128)
                __builtin_prefetch(Vg + (size_t)(tid - 64) * S_LEN + jb + 64, 0, 1);
        }
#ifdef USE_ASYNC_LDS
        if (it + 1 < NSTEP) __builtin_amdgcn_s_wait_asynccnt(2);  // cur done
        else                __builtin_amdgcn_s_wait_asynccnt(0);
#endif
        __syncthreads();   // buffer cur visible to all waves

        const _Float16* tK = sK[cur];
        const _Float16* tV = sV[cur];

        // ---- scores: S = Q @ K^T (2 q-tiles x 2 key-tiles x 2 K-chunks) ----
        v8f sc[2][2];
        #pragma unroll
        for (int mt = 0; mt < 2; ++mt)
            #pragma unroll
            for (int kt = 0; kt < 2; ++kt)
                #pragma unroll
                for (int r = 0; r < 8; ++r) sc[mt][kt][r] = 0.0f;
        #pragma unroll
        for (int kc = 0; kc < 2; ++kc) {
            v16h b0, b1;   // B[kk=channel][nn=key]: lane N = key, K = 16h + j
            const _Float16* kr0 = tK + (size_t)l15 * CH + (kc << 5) + (h << 4);
            const _Float16* kr1 = kr0 + 16 * CH;
            #pragma unroll
            for (int j = 0; j < 16; ++j) { b0[j] = kr0[j]; b1[j] = kr1[j]; }
            #pragma unroll
            for (int mt = 0; mt < 2; ++mt) {
                sc[mt][0] = wmma_f16(aq[mt][kc], b0, sc[mt][0]);
                sc[mt][1] = wmma_f16(aq[mt][kc], b1, sc[mt][1]);
            }
        }

        // ---- register softmax: row (mt, r+8h), cols split over the half ----
        float alpha[2][8];
        #pragma unroll
        for (int mt = 0; mt < 2; ++mt)
            #pragma unroll
            for (int r = 0; r < 8; ++r) {
                float s0v = sc[mt][0][r] * 0.125f;
                float s1v = sc[mt][1][r] * 0.125f;
                float mx = fmaxf(s0v, s1v);
                mx = fmaxf(mx, __shfl_xor(mx, 1, 32));
                mx = fmaxf(mx, __shfl_xor(mx, 2, 32));
                mx = fmaxf(mx, __shfl_xor(mx, 4, 32));
                mx = fmaxf(mx, __shfl_xor(mx, 8, 32));
                float m_new = fmaxf(m_run[mt][r], mx);
                float a0 = __expf(m_run[mt][r] - m_new);
                m_run[mt][r] = m_new;
                alpha[mt][r] = a0;
                float p0 = __expf(s0v - m_new);
                float p1 = __expf(s1v - m_new);
                int row = (mt << 4) + r + (h << 3);
                sP[wv][row * 32 + l15]      = (_Float16)p0;
                sP[wv][row * 32 + 16 + l15] = (_Float16)p1;
                float sum = p0 + p1;
                sum += __shfl_xor(sum, 1, 32);
                sum += __shfl_xor(sum, 2, 32);
                sum += __shfl_xor(sum, 4, 32);
                sum += __shfl_xor(sum, 8, 32);
                l_run[mt][r] = l_run[mt][r] * a0 + sum;
            }

        // rescale accumulators (alpha index matches acc row index exactly)
        #pragma unroll
        for (int mt = 0; mt < 2; ++mt)
            #pragma unroll
            for (int t = 0; t < 4; ++t)
                #pragma unroll
                for (int r = 0; r < 8; ++r) acc[mt][t][r] *= alpha[mt][r];

        // ---- P @ V: P (2 A tiles) from LDS, V^T tiles as B from LDS ----
        v16h ap[2];
        #pragma unroll
        for (int mt = 0; mt < 2; ++mt)
            #pragma unroll
            for (int j = 0; j < 8; ++j) {
                ap[mt][j]     = sP[wv][((mt << 4) + l15) * 32 + (h << 4) + j];
                ap[mt][8 + j] = sP[wv][((mt << 4) + l15) * 32 + 8 + (h << 4) + j];
            }
        #pragma unroll
        for (int t = 0; t < 4; ++t) {
            v16h bvv;   // B[kk=key][nn=channel]: lane N = channel, K = 16h + j
            const _Float16* vr = tV + ((t << 4) + l15) * 32 + (h << 4);
            #pragma unroll
            for (int j = 0; j < 16; ++j) bvv[j] = vr[j];
            #pragma unroll
            for (int mt = 0; mt < 2; ++mt)
                acc[mt][t] = wmma_f16(ap[mt], bvv, acc[mt][t]);
        }
    }

    // final 1/l normalization + de-window store (f16, row-major [16384][512])
    const int head = g & 7;
    const int bw   = g >> 3;
    const int b    = bw >> 2;
    const int win  = bw & 3;
    #pragma unroll
    for (int mt = 0; mt < 2; ++mt) {
        #pragma unroll
        for (int r = 0; r < 8; ++r) {
            float li = 1.0f / l_run[mt][r];
            int s   = q0 + (mt << 4) + r + (h << 3);
            int tt  = s >> 8, rem = s & 255;
            int hh  = ((win >> 1) << 4) | (rem >> 4);
            int ww  = ((win & 1) << 4) | (rem & 15);
            int row = (((b << 3) | tt) << 10) + (hh << 5) + ww;
            #pragma unroll
            for (int t = 0; t < 4; ++t) {
                int col = (head << 6) | ((t << 4) + l15);
                Att[(size_t)row * C_DIM + col] = (_Float16)(acc[mt][t][r] * li);
            }
        }
    }
}

// ---------------------------------------------------------------------------
// Kernel 3: output projection  out = att @ Wo^T + bo  (32x32 per wave, f32 out)
// ---------------------------------------------------------------------------
__global__ __launch_bounds__(32)
void out_proj_kernel(const _Float16* __restrict__ Att,
                     const _Float16* __restrict__ Who, const float* __restrict__ bo,
                     float* __restrict__ out) {
    const int lane = threadIdx.x;
    const int h    = lane >> 4;
    const int l15  = lane & 15;
    const int m0   = blockIdx.x << 5;
    const int n0   = blockIdx.y << 5;

    v8f c[2][2];
    #pragma unroll
    for (int nt = 0; nt < 2; ++nt) {
        float bvv = bo[n0 + (nt << 4) + l15];
        #pragma unroll
        for (int mt = 0; mt < 2; ++mt)
            #pragma unroll
            for (int r = 0; r < 8; ++r) c[mt][nt][r] = bvv;
    }

    for (int k0 = 0; k0 < C_DIM; k0 += 32) {
        v16h a[2];
        #pragma unroll
        for (int mt = 0; mt < 2; ++mt) {
            const _Float16* ar = Att + (size_t)(m0 + (mt << 4) + l15) * C_DIM + k0;
            #pragma unroll
            for (int j = 0; j < 8; ++j) {
                a[mt][j]     = ar[(h << 4) + j];
                a[mt][8 + j] = ar[8 + (h << 4) + j];
            }
        }
        #pragma unroll
        for (int nt = 0; nt < 2; ++nt) {
            v16h b;
            const _Float16* wr =
                Who + (size_t)(n0 + (nt << 4) + l15) * C_DIM + k0 + (h << 4);
            #pragma unroll
            for (int j = 0; j < 16; ++j) b[j] = wr[j];
            #pragma unroll
            for (int mt = 0; mt < 2; ++mt)
                c[mt][nt] = wmma_f16(a[mt], b, c[mt][nt]);
        }
    }

    #pragma unroll
    for (int mt = 0; mt < 2; ++mt)
        #pragma unroll
        for (int r = 0; r < 8; ++r)
            #pragma unroll
            for (int nt = 0; nt < 2; ++nt)
                out[(size_t)(m0 + (mt << 4) + r + (h << 3)) * C_DIM +
                    n0 + (nt << 4) + l15] = c[mt][nt][r];
}

// ---------------------------------------------------------------------------
extern "C" void kernel_launch(void* const* d_in, const int* in_sizes, int n_in,
                              void* d_out, int out_size, void* d_ws, size_t ws_size,
                              hipStream_t stream) {
    const float* x  = (const float*)d_in[0];
    const float* Wq = (const float*)d_in[1];
    const float* bq = (const float*)d_in[2];
    const float* Wk = (const float*)d_in[3];
    const float* bk = (const float*)d_in[4];
    const float* Wv = (const float*)d_in[5];
    const float* bv = (const float*)d_in[6];
    const float* Wo = (const float*)d_in[7];
    const float* bo = (const float*)d_in[8];
    float* out = (float*)d_out;

    char* ws = (char*)d_ws;
    const size_t SEG = (size_t)M_TOTAL * C_DIM * 2;   // 16 MB per f16 tensor
    const size_t WSEG = (size_t)C_DIM * C_DIM * 2;    // 0.5 MB per f16 weight
    _Float16* Qw  = (_Float16*)(ws + 0 * SEG);        // windowed row-major
    _Float16* Kw  = (_Float16*)(ws + 1 * SEG);        // windowed row-major
    _Float16* Vt  = (_Float16*)(ws + 2 * SEG);        // windowed, transposed
    _Float16* Att = (_Float16*)(ws + 3 * SEG);        // de-windowed row-major
    _Float16* Xh  = (_Float16*)(ws + 4 * SEG);        // x in f16
    _Float16* Whq = (_Float16*)(ws + 5 * SEG + 0 * WSEG);
    _Float16* Whk = (_Float16*)(ws + 5 * SEG + 1 * WSEG);
    _Float16* Whv = (_Float16*)(ws + 5 * SEG + 2 * WSEG);
    _Float16* Who = (_Float16*)(ws + 5 * SEG + 3 * WSEG);

    const int NX = M_TOTAL * C_DIM;   // 8388608
    const int NW = C_DIM * C_DIM;     // 262144
    cvt_f32_to_f16_kernel<<<NX / 1024, 256, 0, stream>>>(x,  Xh,  NX);
    cvt_f32_to_f16_kernel<<<NW / 1024, 256, 0, stream>>>(Wq, Whq, NW);
    cvt_f32_to_f16_kernel<<<NW / 1024, 256, 0, stream>>>(Wk, Whk, NW);
    cvt_f32_to_f16_kernel<<<NW / 1024, 256, 0, stream>>>(Wv, Whv, NW);
    cvt_f32_to_f16_kernel<<<NW / 1024, 256, 0, stream>>>(Wo, Who, NW);

    qkv_proj_kernel<<<dim3(M_TOTAL / 32, C_DIM / 32), 32, 0, stream>>>(
        Xh, Whq, bq, Whk, bk, Whv, bv, Qw, Kw, Vt);
    attn_kernel<<<dim3(S_LEN / 256, NGROUP), 256, 0, stream>>>(Qw, Kw, Vt, Att);
    out_proj_kernel<<<dim3(M_TOTAL / 32, C_DIM / 32), 32, 0, stream>>>(
        Att, Who, bo, out);
}
